// LearnableRoIPool_82592221102154
// MI455X (gfx1250) — matrix operand order repaired and checked by reading, not compiled, not run on hardware
//
#include <hip/hip_runtime.h>
#include <hip/hip_bf16.h>

typedef __attribute__((ext_vector_type(16))) _Float16 v16h;
typedef __attribute__((ext_vector_type(8)))  float    v8f;

// Problem sizes (fixed by the reference)
static constexpr int Bc = 4, Cc = 256, Hc = 64, Wc = 48, Nroi = 32;
static constexpr int Dd = 512, OUTD = 256, Kk = 49;
static constexpr int HW = Hc * Wc;      // 3072
static constexpr int NK = Nroi * Kk;    // 1568

// Workspace layout (in floats)
static constexpr size_t OFF_EPS  = 0;                                  // 32 x 2048
static constexpr size_t OFF_V    = OFF_EPS  + (size_t)Nroi * 4 * Dd;   // 32 x 512
static constexpr size_t OFF_BVEC = OFF_V    + (size_t)Nroi * Dd;       // 32 x 12544
static constexpr size_t OFF_EX   = OFF_BVEC + (size_t)Nroi * Kk * OUTD;// 64 x 512
static constexpr size_t OFF_EY   = OFF_EX   + (size_t)Hc * Dd;         // 48 x 512
static constexpr size_t OFF_IMGH = OFF_EY   + (size_t)Wc * Dd;         // 64 x 256
static constexpr size_t OFF_IMGW = OFF_IMGH + (size_t)Hc * OUTD;       // 48 x 256
static constexpr size_t OFF_GEOH = OFF_IMGW + (size_t)Wc * OUTD;       // 1568 x 64
static constexpr size_t OFF_GEOW = OFF_GEOH + (size_t)NK * Hc;         // 1568 x 48
static constexpr size_t OFF_APP  = OFF_GEOW + (size_t)NK * Wc;         // 4 x 49 x 3072
static constexpr size_t OFF_INV  = OFF_APP  + (size_t)Bc * Kk * HW;    // 32 ints
static constexpr size_t WS_FLOATS = OFF_INV + 32;

// ---------------------------------------------------------------- embeddings
__device__ __forceinline__ float inv_denom(int d) {
  float e = (d < 256) ? (2.0f * (float)d + 1.0f) * (1.0f / 512.0f)
                      : (2.0f * (float)(d - 256)) * (1.0f / 512.0f);
  return __expf(-e * 6.90775527898214f);   // 1000^-e
}

__global__ void embed_rois_kernel(const float* __restrict__ rois, float* __restrict__ eps) {
  int i = blockIdx.x * blockDim.x + threadIdx.x;
  if (i >= Nroi * 4 * Dd) return;
  int d  = i & (Dd - 1);
  int nj = i / Dd;
  int j  = nj & 3;
  int n  = nj >> 2;
  float z = rois[n * 5 + 1 + j];
  float t = z * inv_denom(d);
  eps[i] = (d < 256) ? __sinf(t) : __cosf(t);
}

__global__ void embed_axes_kernel(float* __restrict__ ex, float* __restrict__ ey) {
  int i = blockIdx.x * blockDim.x + threadIdx.x;
  if (i >= (Hc + Wc) * Dd) return;
  int d = i & (Dd - 1);
  int a = i / Dd;
  float z = (a < Hc) ? (float)a : (float)(a - Hc);
  float t = z * inv_denom(d);
  float v = (d < 256) ? __sinf(t) : __cosf(t);
  if (a < Hc) ex[a * Dd + d] = v;
  else        ey[(a - Hc) * Dd + d] = v;
}

// ---------------------------------------------------------------- WMMA GEMM
// C(MxN, row-major ldc) = A(MxK, row-major lda) * B^T, B element (n,k) at
// B[n*bs_n + k*bs_k].  One wave32 per 16x16 tile, v_wmma_f32_16x16x32_f16,
// f32 accumulation.  Requirements imposed by the launches below:
//   - K multiple of 32, N multiple of 16 (so no column guard at all)
//   - MGUARD=true only when M is not a multiple of 16 (row clamp on loads,
//     row guard on stores, uniform only within each lane half)
template <bool BCONTIG, bool MGUARD>
__global__ __launch_bounds__(32) void gemm_wmma_kernel(
    const float* __restrict__ A, int lda,
    const float* __restrict__ B, int bs_n, int bs_k,
    float* __restrict__ C, int ldc,
    int M, int N, int Ktot)
{
  const int lane = threadIdx.x & 31;
  const int m16  = lane & 15;
  const int hi   = lane >> 4;
  const int tN = blockIdx.x, tM = blockIdx.y;
  const int row0 = tM * 16 + m16;                       // logical A row
  const int rowc = MGUARD ? min(row0, M - 1) : row0;    // clamped for loads
  const int col  = tN * 16 + m16;                       // B/C column (< N always)

  // A fragment: lane-half hi holds K = hi*16+[0..15] (matches B).
  const float* Arow = A + rowc * lda + hi * 16;
  // B (KxN) fragment: lane-half hi holds K = hi*16+[0..15].
  const float* Bcol = BCONTIG ? (B + col * bs_n + hi * 16)
                              : (B + col * bs_n + (hi * 16) * bs_k);

  v8f acc = {};
  for (int k0 = 0; k0 < Ktot; k0 += 32) {
    v16h a, b;
    {
      float4 a0 = *(const float4*)(Arow + k0);
      float4 a1 = *(const float4*)(Arow + k0 + 4);
      float4 a2 = *(const float4*)(Arow + k0 + 8);
      float4 a3 = *(const float4*)(Arow + k0 + 12);
      a[0]=(_Float16)a0.x; a[1]=(_Float16)a0.y; a[2]=(_Float16)a0.z; a[3]=(_Float16)a0.w;
      a[4]=(_Float16)a1.x; a[5]=(_Float16)a1.y; a[6]=(_Float16)a1.z; a[7]=(_Float16)a1.w;
      a[8]=(_Float16)a2.x; a[9]=(_Float16)a2.y; a[10]=(_Float16)a2.z; a[11]=(_Float16)a2.w;
      a[12]=(_Float16)a3.x; a[13]=(_Float16)a3.y; a[14]=(_Float16)a3.z; a[15]=(_Float16)a3.w;
    }
    if (BCONTIG) {
      float4 b0 = *(const float4*)(Bcol + k0);
      float4 b1 = *(const float4*)(Bcol + k0 + 4);
      float4 b2 = *(const float4*)(Bcol + k0 + 8);
      float4 b3 = *(const float4*)(Bcol + k0 + 12);
      b[0]=(_Float16)b0.x; b[1]=(_Float16)b0.y; b[2]=(_Float16)b0.z; b[3]=(_Float16)b0.w;
      b[4]=(_Float16)b1.x; b[5]=(_Float16)b1.y; b[6]=(_Float16)b1.z; b[7]=(_Float16)b1.w;
      b[8]=(_Float16)b2.x; b[9]=(_Float16)b2.y; b[10]=(_Float16)b2.z; b[11]=(_Float16)b2.w;
      b[12]=(_Float16)b3.x; b[13]=(_Float16)b3.y; b[14]=(_Float16)b3.z; b[15]=(_Float16)b3.w;
    } else {
#pragma unroll
      for (int e = 0; e < 16; ++e)
        b[e] = (_Float16)Bcol[(k0 + e) * bs_k];
    }
    acc = __builtin_amdgcn_wmma_f32_16x16x32_f16(
        /*neg_a=*/false, a, /*neg_b=*/false, b,
        /*c_mod=*/(short)0, acc, /*reuse_a=*/false, /*reuse_b=*/false);
  }

  // C/D layout: VGPR r -> row tM*16 + hi*8 + r, column = col.
  const int rbase = tM * 16 + hi * 8;
  float* Crow = C + rbase * ldc + col;
#pragma unroll
  for (int r = 0; r < 8; ++r) {
    if (!MGUARD || (rbase + r) < M)      // uniform only within a lane half
      Crow[r * ldc] = acc[r];
  }
}

// ---------------------------------------------------------------- permutation
__global__ void order_kernel(const float* __restrict__ rois, int* __restrict__ inv) {
  if (blockIdx.x != 0 || threadIdx.x != 0) return;
  for (int n = 0; n < Nroi; ++n) {
    float zn = rois[n * 5];
    int r = 0;
    for (int m = 0; m < Nroi; ++m) {
      float zm = rois[m * 5];
      if (zm < zn || (zm == zn && m < n)) ++r;
    }
    inv[n] = r;
  }
}

// ---------------------------------------------------------------- fused tail
__global__ __launch_bounds__(256) void fused_softmax_kernel(
    const float* __restrict__ app, const float* __restrict__ geoH,
    const float* __restrict__ geoW, const float* __restrict__ conv_b,
    const int* __restrict__ inv, float* __restrict__ out)
{
  int i = blockIdx.x * blockDim.x + threadIdx.x;
  if (i >= Nroi * Bc * HW) return;
  int hw = i % HW;
  int nb = i / HW;
  int b  = nb & (Bc - 1);
  int n  = nb / Bc;
  int h  = hw / Wc;
  int w  = hw - h * Wc;

  const float* gH = geoH + n * Kk * Hc + h;
  const float* gW = geoW + n * Kk * Wc + w;
  const float* ap = app  + b * Kk * HW + hw;

  float l[Kk];
  float mx = -3.4e38f;
#pragma unroll
  for (int k = 0; k < Kk; ++k) {
    float v = ap[k * HW] + conv_b[k] + gH[k * Hc] + gW[k * Wc];
    l[k] = v;
    mx = fmaxf(mx, v);
  }
  float s = 0.0f;
#pragma unroll
  for (int k = 0; k < Kk; ++k) { float e = __expf(l[k] - mx); l[k] = e; s += e; }
  float r = 1.0f / s;
  int p = inv[n];
  float* o = out + ((size_t)p * Bc + b) * Kk * HW + hw;
#pragma unroll
  for (int k = 0; k < Kk; ++k) o[(size_t)k * HW] = l[k] * r;
}

// ---------------------------------------------------------------- launcher
extern "C" void kernel_launch(void* const* d_in, const int* in_sizes, int n_in,
                              void* d_out, int out_size, void* d_ws, size_t ws_size,
                              hipStream_t stream) {
  (void)in_sizes; (void)n_in; (void)out_size;
  const float* x      = (const float*)d_in[0];  // (4,256,64,48)
  const float* rois   = (const float*)d_in[1];  // (32,5)
  const float* V_box  = (const float*)d_in[2];  // (512,2048)
  const float* W_box  = (const float*)d_in[3];  // (49,256,512)
  const float* W_im   = (const float*)d_in[4];  // (256,1024)
  const float* conv_w = (const float*)d_in[5];  // (49,256)
  const float* conv_b = (const float*)d_in[6];  // (49,)
  float* out = (float*)d_out;
  float* ws  = (float*)d_ws;
  if (ws_size < WS_FLOATS * sizeof(float)) return;

  float* eps  = ws + OFF_EPS;
  float* v    = ws + OFF_V;
  float* bvec = ws + OFF_BVEC;
  float* ex   = ws + OFF_EX;
  float* ey   = ws + OFF_EY;
  float* imgH = ws + OFF_IMGH;
  float* imgW = ws + OFF_IMGW;
  float* geoH = ws + OFF_GEOH;
  float* geoW = ws + OFF_GEOW;
  float* app  = ws + OFF_APP;
  int*   inv  = (int*)(ws + OFF_INV);

  embed_rois_kernel<<<(Nroi * 4 * Dd + 255) / 256, 256, 0, stream>>>(rois, eps);
  embed_axes_kernel<<<((Hc + Wc) * Dd + 255) / 256, 256, 0, stream>>>(ex, ey);

  // v[n,d] = sum_f eps[n,f] * V_box[d,f]           (32 x 512, K=2048)
  gemm_wmma_kernel<true, false><<<dim3(Dd / 16, Nroi / 16), 32, 0, stream>>>(
      eps, 4 * Dd, V_box, 4 * Dd, 1, v, Dd, Nroi, Dd, 4 * Dd);

  // bvec[n, k*256+o] = sum_d v[n,d] * W_box[k,o,d] (32 x 12544, K=512)
  gemm_wmma_kernel<true, false><<<dim3((Kk * OUTD) / 16, Nroi / 16), 32, 0, stream>>>(
      v, Dd, W_box, Dd, 1, bvec, Kk * OUTD, Nroi, Kk * OUTD, Dd);

  // imgH[h,o] = sum_d ex[h,d] * W_im[o,d]          (64 x 256, K=512)
  gemm_wmma_kernel<true, false><<<dim3(OUTD / 16, Hc / 16), 32, 0, stream>>>(
      ex, Dd, W_im, 2 * Dd, 1, imgH, OUTD, Hc, OUTD, Dd);
  // imgW[w,o] = sum_d ey[w,d] * W_im[o, 512+d]     (48 x 256, K=512)
  gemm_wmma_kernel<true, false><<<dim3(OUTD / 16, Wc / 16), 32, 0, stream>>>(
      ey, Dd, W_im + Dd, 2 * Dd, 1, imgW, OUTD, Wc, OUTD, Dd);

  // geoH[nk,h] = sum_o bvec[nk,o] * imgH[h,o]      (1568 x 64, K=256)
  gemm_wmma_kernel<true, false><<<dim3(Hc / 16, NK / 16), 32, 0, stream>>>(
      bvec, OUTD, imgH, OUTD, 1, geoH, Hc, NK, Hc, OUTD);
  // geoW[nk,w] = sum_o bvec[nk,o] * imgW[w,o]      (1568 x 48, K=256)
  gemm_wmma_kernel<true, false><<<dim3(Wc / 16, NK / 16), 32, 0, stream>>>(
      bvec, OUTD, imgW, OUTD, 1, geoW, Wc, NK, Wc, OUTD);

  // app[b,k,hw] = sum_c conv_w[k,c] * x[b,c,hw]    (49 x 3072 per batch, K=256)
  // B = x[b] with element (hw, c) at x[b, c*HW + hw]  -> strided (BCONTIG=false)
  for (int b = 0; b < Bc; ++b) {
    gemm_wmma_kernel<false, true><<<dim3(HW / 16, (Kk + 15) / 16), 32, 0, stream>>>(
        conv_w, Cc, x + (size_t)b * Cc * HW, 1, HW,
        app + (size_t)b * Kk * HW, HW, Kk, HW, Cc);
  }

  order_kernel<<<1, 32, 0, stream>>>(rois, inv);

  fused_softmax_kernel<<<(Nroi * Bc * HW + 255) / 256, 256, 0, stream>>>(
      app, geoH, geoW, conv_b, inv, out);
}
